// BiaffineLabelAttention_66279935312114
// MI455X (gfx1250) — matrix operand order repaired and checked by reading, not compiled, not run on hardware
//
#include <hip/hip_runtime.h>
#include <stdint.h>

typedef __attribute__((ext_vector_type(16))) __bf16 v16bf;
typedef __attribute__((ext_vector_type(8)))  float  v8f;

// Problem constants (B,S,D,L) = (16,256,768,32)
#define BSZ 16
#define SSZ 256
#define DSZ 768
#define LSZ 32

#define TM  256          // output tile rows per workgroup (full sequence)
#define TN  128          // output tile cols per workgroup
#define TK  32           // K per LDS stage == WMMA K
#define LDT 40           // padded K-stride (elements) in LDS tiles (bank spread)
#define NKT (DSZ / TK)   // 24 k-stages

struct alignas(16) F4 { float x, y, z, w; };
struct alignas(8)  F2 { float x, y; };

// ---------------------------------------------------------------------------
// Kernel 1: f32 -> bf16 bulk convert (head and dep), so both GEMM operands
// are pure async byte copies in the main loop.
// ---------------------------------------------------------------------------
__global__ __launch_bounds__(256)
void cvt_bf16(const float* __restrict__ x, __bf16* __restrict__ y) {
  size_t i = ((size_t)blockIdx.x * 256 + threadIdx.x) * 4;
  F4 v = *(const F4*)(x + i);
  union { __bf16 h[4]; F2 q; } p;
  p.h[0] = (__bf16)v.x; p.h[1] = (__bf16)v.y;
  p.h[2] = (__bf16)v.z; p.h[3] = (__bf16)v.w;
  *(F2*)(y + i) = p.q;
}

// ---------------------------------------------------------------------------
// Kernel 2: t2_h[b,l,i] = dot(W[l,0:D], head[b,i,:])
//           t2_d[b,l,o] = dot(W[l,D:2D], dep[b,o,:])
// ---------------------------------------------------------------------------
__global__ __launch_bounds__(256)
void t2_kernel(const float* __restrict__ head, const float* __restrict__ dep,
               const float* __restrict__ W,
               float* __restrict__ t2h, float* __restrict__ t2d) {
  const int l = blockIdx.x, b = blockIdx.y, i = threadIdx.x;
  __shared__ float wh[DSZ], wd[DSZ];
  for (int k = i; k < DSZ; k += 256) {
    wh[k] = W[(size_t)l * 2 * DSZ + k];
    wd[k] = W[(size_t)l * 2 * DSZ + DSZ + k];
  }
  __syncthreads();
  const F4* hp  = (const F4*)(head + ((size_t)b * SSZ + i) * DSZ);
  const F4* dp  = (const F4*)(dep  + ((size_t)b * SSZ + i) * DSZ);
  const F4* whp = (const F4*)wh;
  const F4* wdp = (const F4*)wd;
  float sh = 0.f, sd = 0.f;
#pragma unroll 4
  for (int k = 0; k < DSZ / 4; ++k) {
    F4 h = hp[k], d = dp[k], a = whp[k], c = wdp[k];
    sh += h.x * a.x + h.y * a.y + h.z * a.z + h.w * a.w;
    sd += d.x * c.x + d.y * c.y + d.z * c.z + d.w * c.w;
  }
  const size_t o = ((size_t)b * LSZ + l) * SSZ + i;
  t2h[o] = sh; t2d[o] = sd;
}

// ---------------------------------------------------------------------------
// Kernel 3: main biaffine GEMM.
//   C[b,l,i,o] = sum_d head[b,i,d]*U[l,d]*dep[b,o,d]  (bf16 WMMA, f32 acc)
//              + t2h[b,l,i] + t2d[b,l,o] + bias[l]
// 8 waves per workgroup, each owning a 64x64 register tile (16 WMMA frags).
// Both A (head_bf16) and B (dep_bf16) tiles are DMA'd into double-buffered
// LDS via global_load_async_to_lds_b128; the per-label U scale is applied to
// the A fragments in registers with packed bf16 multiplies (U frag is an LDS
// broadcast shared by all rows).
// ---------------------------------------------------------------------------
__global__ __launch_bounds__(256)
void biaffine_wmma(const __bf16* __restrict__ headbf,
                   const __bf16* __restrict__ depbf,
                   const float* __restrict__ U,
                   const float* __restrict__ t2h,
                   const float* __restrict__ t2d,
                   const float* __restrict__ bias,
                   float* __restrict__ out) {
  __shared__ __bf16 As[2][TM * LDT];   // 40 KB
  __shared__ __bf16 Bs[2][TN * LDT];   // 20 KB
  __shared__ __bf16 Ubf[DSZ];          // 1.5 KB
  __shared__ float  t2hs[TM];          // 1 KB
  __shared__ float  t2ds[TN];          // 0.5 KB

  const int tid    = threadIdx.x;
  const int lane   = tid & 31;
  const int wave   = tid >> 5;
  const int wm     = (wave & 3) * 64;   // wave row base in tile
  const int wn     = (wave >> 2) * 64;  // wave col base in tile
  const int laneLo = lane & 15;
  const int hiH    = lane >> 4;         // 0 or 1 (half-wave)

  const int l  = blockIdx.y;
  const int b  = blockIdx.z;
  const int n0 = blockIdx.x * TN;       // m0 == 0 (TM spans full S)

  // ---- prologue LDS fills -------------------------------------------------
  for (int k = tid; k < DSZ; k += 256) Ubf[k] = (__bf16)U[(size_t)l * DSZ + k];
  const size_t t2base = ((size_t)b * LSZ + l) * SSZ;
  t2hs[tid] = t2h[t2base + tid];
  if (tid < TN) t2ds[tid] = t2d[t2base + n0 + tid];

  const uint64_t hbase = (uint64_t)(uintptr_t)(headbf + (size_t)b * SSZ * DSZ);
  const uint64_t dbase = (uint64_t)(uintptr_t)(depbf  + (size_t)b * SSZ * DSZ);

  // A stage: 256 rows x 32 k bf16 = 16 KB; one row (64 B = 4x b128) / thread.
  // The instruction offset applies to BOTH the global and LDS address
  // (ISA 08 §4.4), so one (lds,voff) pair covers all four chunks.
  auto loadA = [&](int st, int k0) {
    uint32_t ldsa = (uint32_t)(uintptr_t)(const void*)&As[st][tid * LDT];
    uint32_t goff = (uint32_t)((tid * DSZ + k0) * 2);
    asm volatile(
        "global_load_async_to_lds_b128 %0, %1, %2\n\t"
        "global_load_async_to_lds_b128 %0, %1, %2 offset:16\n\t"
        "global_load_async_to_lds_b128 %0, %1, %2 offset:32\n\t"
        "global_load_async_to_lds_b128 %0, %1, %2 offset:48"
        :: "v"(ldsa), "v"(goff), "s"(hbase) : "memory");
  };
  // B stage: 128 rows x 32 k bf16 = 8 KB; half a row (2x b128) / thread.
  const int crow  = tid >> 1;
  const int chalf = (tid & 1) * 16;
  auto loadB = [&](int st, int k0) {
    uint32_t ldsa = (uint32_t)(uintptr_t)(const void*)&Bs[st][crow * LDT + chalf];
    uint32_t goff = (uint32_t)(((n0 + crow) * DSZ + k0 + chalf) * 2);
    asm volatile(
        "global_load_async_to_lds_b128 %0, %1, %2\n\t"
        "global_load_async_to_lds_b128 %0, %1, %2 offset:16"
        :: "v"(ldsa), "v"(goff), "s"(dbase) : "memory");
  };

  v8f acc[4][4] = {};

  auto compute = [&](int st, int k0) {
    union V { v16bf v; F4 q[2]; };
    V u, a, bb[4];
    // U frag mirrors the A-frag K layout, identical for every row ->
    // LDS broadcast reads, shared by all 4 A frags.
    const __bf16* up = &Ubf[k0 + hiH * 8];
    u.q[0] = *(const F4*)up;         // K = hiH*8 .. +7
    u.q[1] = *(const F4*)(up + 16);  // K = 16+hiH*8 .. +7
    // B frags (same K layout as A): col=laneLo, K = hiH*8..+7 and 16+hiH*8..+7
#pragma unroll
    for (int j = 0; j < 4; ++j) {
      const __bf16* p = &Bs[st][(wn + j * 16 + laneLo) * LDT + hiH * 8];
      bb[j].q[0] = *(const F4*)p;
      bb[j].q[1] = *(const F4*)(p + 16);
    }
    // A frags (16-bit 16x32 layout), scaled by U in registers (v_pk_mul_bf16)
#pragma unroll
    for (int i = 0; i < 4; ++i) {
      const __bf16* p = &As[st][(wm + i * 16 + laneLo) * LDT + hiH * 8];
      a.q[0] = *(const F4*)p;
      a.q[1] = *(const F4*)(p + 16);
      a.v = a.v * u.v;
#pragma unroll
      for (int j = 0; j < 4; ++j)
        acc[i][j] = __builtin_amdgcn_wmma_f32_16x16x32_bf16(
            false, a.v, false, bb[j].v, (short)0, acc[i][j], false, false);
    }
  };

  // ---- software-pipelined main loop (double-buffered LDS, async DMA) ------
  loadA(0, 0);
  loadB(0, 0);
  asm volatile("s_wait_asynccnt 0" ::: "memory");
  __syncthreads();

  for (int kt = 0; kt < NKT; ++kt) {
    const int st = kt & 1;
    if (kt + 1 < NKT) {
      loadA(st ^ 1, (kt + 1) * TK);
      loadB(st ^ 1, (kt + 1) * TK);
    }
    compute(st, kt * TK);
    if (kt + 1 < NKT)
      asm volatile("s_wait_asynccnt 0" ::: "memory");
    __syncthreads();
  }

  // ---- epilogue: + t2h[row] + t2d[col] + bias[l] --------------------------
  const float bv = bias[l];
  float* outp = out + (((size_t)b * LSZ + l) * SSZ) * SSZ;
#pragma unroll
  for (int i = 0; i < 4; ++i) {
    const int rowBase = wm + i * 16 + hiH * 8;   // C frag: VGPR r -> M=r(+8 hi)
#pragma unroll
    for (int j = 0; j < 4; ++j) {
      const int colL = wn + j * 16 + laneLo;     // C frag: lane -> N
      const float td = t2ds[colL] + bv;
#pragma unroll
      for (int r = 0; r < 8; ++r) {
        const int rowL = rowBase + r;
        outp[(size_t)rowL * SSZ + (n0 + colL)] =
            acc[i][j][r] + t2hs[rowL] + td;
      }
    }
  }
}

// ---------------------------------------------------------------------------
extern "C" void kernel_launch(void* const* d_in, const int* in_sizes, int n_in,
                              void* d_out, int out_size, void* d_ws, size_t ws_size,
                              hipStream_t stream) {
  const float* head = (const float*)d_in[0];
  const float* dep  = (const float*)d_in[1];
  const float* U    = (const float*)d_in[2];
  const float* W    = (const float*)d_in[3];
  const float* bias = (const float*)d_in[4];
  float* out = (float*)d_out;

  // workspace: head_bf16 | dep_bf16 | t2h | t2d  (~13.6 MB)
  const size_t nElem = (size_t)BSZ * SSZ * DSZ;
  __bf16* headbf = (__bf16*)d_ws;
  __bf16* depbf  = headbf + nElem;
  float*  t2h = (float*)(depbf + nElem);
  float*  t2d = t2h + (size_t)BSZ * LSZ * SSZ;

  cvt_bf16<<<nElem / (256 * 4), 256, 0, stream>>>(head, headbf);
  cvt_bf16<<<nElem / (256 * 4), 256, 0, stream>>>(dep, depbf);
  t2_kernel<<<dim3(LSZ, BSZ), 256, 0, stream>>>(head, dep, W, t2h, t2d);
  biaffine_wmma<<<dim3(SSZ / TN, LSZ, BSZ), 256, 0, stream>>>(
      headbf, depbf, U, t2h, t2d, bias, out);
}
